// WaveletFPN_Downsample_31653908971648
// MI455X (gfx1250) — matrix operand order repaired and checked by reading, not compiled, not run on hardware
//
#include <hip/hip_runtime.h>
#include <math.h>
#include <stdint.h>

typedef __attribute__((ext_vector_type(16))) _Float16 v16h;
typedef __attribute__((ext_vector_type(8)))  _Float16 v8h;
typedef __attribute__((ext_vector_type(2)))  _Float16 v2h;
typedef __attribute__((ext_vector_type(8)))  float    v8f;

#define BATCH 8
#define C1    256
#define Hh    128
#define Ww    128
#define SPAT  4096              // 64*64
#define KC    1024              // fused channels (256 LL + 768 HF)
#define OC    512
#define MIDC  48
#define HFC   768

// WGP-scope prefetch (scope 0 pulls into ALL cache levels, ISA §10.5).
// __builtin_prefetch can only reach SE/SYS scope, which skips the WGP cache.
__device__ __forceinline__ void prefetch_wgp(const void* p) {
  asm volatile("global_prefetch_b8 %0, off" :: "v"((unsigned long long)(uintptr_t)p));
}

// ---------------------------------------------------------------- utilities
__global__ void k_zero(float* __restrict__ p, int n) {
  int i = blockIdx.x * blockDim.x + threadIdx.x;
  if (i < n) p[i] = 0.0f;
}

__device__ __forceinline__ void st8h(_Float16* dst, const _Float16* src) {
  v8h v;
  #pragma unroll
  for (int i = 0; i < 8; ++i) v[i] = src[i];
  *(v8h*)dst = v;   // 16B store
}

// ---------------------------------------------------------------- kernel 1
// Haar wavelet. Each thread owns TWO adjacent output pixels so every lane
// reads one contiguous float4 per input row (fully coalesced 512B/wave).
// fused tensor written transposed [b][s][c] as f16 in 16B v8h chunks.
// HF channel sums (pre-scale) accumulated via wave32 shuffle + f32 atomics.
__global__ void __launch_bounds__(256)
k_wavelet(const float* __restrict__ x,
          _Float16* __restrict__ fusedT,
          float* __restrict__ hfsum) {
  int blk   = blockIdx.x;          // b*8 + chunk
  int b     = blk >> 3;
  int chunk = blk & 7;
  int t     = threadIdx.x;
  int p     = chunk * 256 + t;     // pixel-pair index, 0..2047
  int h2    = p >> 5;              // 0..63
  int q     = p & 31;              // pair column: covers w2 = 2q, 2q+1
  int lane  = t & 31;

  const float* xb = x + (size_t)b * C1 * Hh * Ww;
  _Float16* frowA = fusedT + ((size_t)(b * SPAT + h2 * 64 + 2 * q)) * KC;
  _Float16* frowB = frowA + KC;

  for (int c0 = 0; c0 < C1; c0 += 8) {
    _Float16 llA[8], llB[8], hfA[24], hfB[24];
    #pragma unroll
    for (int cc = 0; cc < 8; ++cc) {
      int c = c0 + cc;
      const float* xc = xb + (size_t)c * Hh * Ww + (size_t)(2 * h2) * Ww + 4 * q;
      float4 r0 = *(const float4*)(xc);
      float4 r1 = *(const float4*)(xc + Ww);
      // pixel A (w2 = 2q)
      float LLa = 0.5f * (r0.x + r0.y + r1.x + r1.y);
      float LHa = 0.5f * (r0.x + r0.y - r1.x - r1.y);
      float HLa = 0.5f * (r0.x - r0.y + r1.x - r1.y);
      float HHa = 0.5f * (r0.x - r0.y - r1.x + r1.y);
      // pixel B (w2 = 2q+1)
      float LLb = 0.5f * (r0.z + r0.w + r1.z + r1.w);
      float LHb = 0.5f * (r0.z + r0.w - r1.z - r1.w);
      float HLb = 0.5f * (r0.z - r0.w + r1.z - r1.w);
      float HHb = 0.5f * (r0.z - r0.w - r1.z + r1.w);

      llA[cc] = (_Float16)LLa;  llB[cc] = (_Float16)LLb;
      hfA[3*cc+0] = (_Float16)LHa; hfA[3*cc+1] = (_Float16)HLa; hfA[3*cc+2] = (_Float16)HHa;
      hfB[3*cc+0] = (_Float16)LHb; hfB[3*cc+1] = (_Float16)HLb; hfB[3*cc+2] = (_Float16)HHb;

      // wave32 reduce the 64 pixels this wave owns for channel sums
      float sLH = LHa + LHb, sHL = HLa + HLb, sHH = HHa + HHb;
      #pragma unroll
      for (int off = 16; off > 0; off >>= 1) {
        sLH += __shfl_down(sLH, off, 32);
        sHL += __shfl_down(sHL, off, 32);
        sHH += __shfl_down(sHH, off, 32);
      }
      if (lane == 0) {
        atomicAdd(&hfsum[b * HFC + 3 * c + 0], sLH);
        atomicAdd(&hfsum[b * HFC + 3 * c + 1], sHL);
        atomicAdd(&hfsum[b * HFC + 3 * c + 2], sHH);
      }
    }
    // 16B vector stores: LL block + three HF blocks per row
    st8h(frowA + c0, llA);
    st8h(frowB + c0, llB);
    #pragma unroll
    for (int j = 0; j < 3; ++j) {
      st8h(frowA + C1 + 3 * c0 + 8 * j, hfA + 8 * j);
      st8h(frowB + C1 + 3 * c0 + 8 * j, hfB + 8 * j);
    }
  }
}

// ---------------------------------------------------------------- kernel 2
// Squeeze-excite MLP. hfsum (24KB) staged into LDS with the CDNA5 async
// path: GLOBAL_LOAD_ASYNC_TO_LDS_B128 + s_wait_asynccnt (ASYNCcnt), then
// barrier. The 1/4096 mean scale is linear -> folded into the first GEMV.
__global__ void __launch_bounds__(256)
k_se(const float* __restrict__ hfsum,
     const float* __restrict__ w1,   // (48,768)
     const float* __restrict__ w2,   // (768,48)
     float* __restrict__ y) {        // (8,768)
  __shared__ float ybar[BATCH * HFC];   // 24 KB (raw sums)
  __shared__ float y1s[BATCH * MIDC];   // 1.5 KB
  int t = threadIdx.x;

  // async copy: 6144 floats = 1536 x b128, 6 per thread
  unsigned lbase = (unsigned)(uintptr_t)(&ybar[0]);  // low 32 bits = LDS offset
  for (int i = t; i < (BATCH * HFC) / 4; i += 256) {
    unsigned ldsoff = lbase + (unsigned)i * 16u;
    unsigned long long ga = (unsigned long long)(uintptr_t)(hfsum + i * 4);
    asm volatile("global_load_async_to_lds_b128 %0, %1, off"
                 :: "v"(ldsoff), "v"(ga) : "memory");
  }
  asm volatile("s_wait_asynccnt 0x0" ::: "memory");
  __syncthreads();

  for (int i = t; i < BATCH * MIDC; i += 256) {
    int b = i / MIDC, m = i % MIDC;
    float acc = 0.0f;
    const float* wr = w1 + (size_t)m * HFC;
    const float* yb = ybar + b * HFC;
    for (int c = 0; c < HFC; ++c) acc += yb[c] * wr[c];
    y1s[i] = fmaxf(acc * (1.0f / (float)SPAT), 0.0f);   // fold mean scale
  }
  __syncthreads();

  for (int i = t; i < BATCH * HFC; i += 256) {
    int b = i / HFC, c = i % HFC;
    float acc = 0.0f;
    const float* wr = w2 + (size_t)c * MIDC;
    const float* yb = y1s + b * MIDC;
    #pragma unroll
    for (int m = 0; m < MIDC; ++m) acc += yb[m] * wr[m];
    y[i] = 1.0f / (1.0f + __expf(-acc));
  }
}

// ---------------------------------------------------------------- kernel 3
// Fold SE scale into per-batch f16 conv weights: w_eff[b][o][c].
// Two elements per thread -> 8B f32 loads, 4B f16 stores.
__global__ void k_weff(const float* __restrict__ conv_w,  // (512,1024)
                       const float* __restrict__ y,       // (8,768)
                       _Float16* __restrict__ weff) {     // (8,512,1024)
  int i2  = blockIdx.x * 256 + threadIdx.x;               // pair index
  int idx = i2 * 2;                                       // < 8*512*1024
  int b   = idx >> 19;
  int rem = idx & ((OC * KC) - 1);
  int c   = rem & (KC - 1);                               // even
  float2 w = *(const float2*)(conv_w + rem);
  float s0 = (c     < C1) ? 1.0f : y[b * HFC + (c     - C1)];
  float s1 = (c + 1 < C1) ? 1.0f : y[b * HFC + (c + 1 - C1)];
  v2h h; h.x = (_Float16)(w.x * s0); h.y = (_Float16)(w.y * s1);
  *(v2h*)(weff + idx) = h;
}

// ---------------------------------------------------------------- kernel 4
// WMMA GEMM per batch: z[b,o,s] = sum_c weff[b,o,c]*fusedT[b,s,c],
// fused BN + SiLU epilogue.  Block = 64(M)x64(N) tile, 8 waves (4x2),
// wave = 16x32 (two v_wmma_f32_16x16x32_f16 accumulators), K-step = 32.
__global__ void __launch_bounds__(256)
k_gemm_bn_silu(const _Float16* __restrict__ weff,
               const _Float16* __restrict__ fusedT,
               const float* __restrict__ gamma,
               const float* __restrict__ beta,
               const float* __restrict__ mean,
               const float* __restrict__ var,
               float* __restrict__ out) {
  int blk  = blockIdx.x;          // ((b*8)+mblk)*64 + nblk
  int nblk = blk & 63;
  int mblk = (blk >> 6) & 7;
  int b    = blk >> 9;
  int t    = threadIdx.x;
  int wave = t >> 5;
  int lane = t & 31;
  int wm   = wave & 3;            // 4 waves along M
  int wn   = wave >> 2;           // 2 waves along N
  int m0   = mblk * 64 + wm * 16;
  int n0   = nblk * 64 + wn * 32;
  int lmod = lane & 15;
  int lhi  = lane >> 4;

  // Per-lane rows (K-contiguous): same K order for A and B (not the ISA table order)
  // A 16x32 f16: lane m=l%16; halves h   -> K = lhi*16+h (one 32B load)
  // B 32x16 f16: lane n=l%16; halves h   -> K = lhi*16+h (one 32B load)
  const _Float16* Arow = weff   + ((size_t)b * OC + (m0 + lmod)) * KC;
  const _Float16* B0   = fusedT + ((size_t)(b * SPAT + n0 + lmod)) * KC;
  const _Float16* B1   = B0 + (size_t)16 * KC;

  v8f acc0 = {};
  v8f acc1 = {};

  #pragma unroll
  for (int k0 = 0; k0 < KC; k0 += 32) {
    if ((k0 & 127) == 0 && k0 + 128 < KC) {   // folds statically (full unroll)
      // WGP-scope prefetch of the next 128-half K-chunk of each lane's row:
      // scope 0 pulls into all cache levels incl. the WGP cache (§10.5).
      prefetch_wgp(Arow + k0 + 128 + lhi * 8);
      prefetch_wgp(B0   + k0 + 128 + lhi * 16);
      prefetch_wgp(B1   + k0 + 128 + lhi * 16);
    }
    v16h a = *(const v16h*)(Arow + k0 + lhi * 16);

    v16h bf0 = *(const v16h*)(B0 + k0 + lhi * 16);
    v16h bf1 = *(const v16h*)(B1 + k0 + lhi * 16);

    acc0 = __builtin_amdgcn_wmma_f32_16x16x32_f16(
        false, a, false, bf0, (short)0, acc0, false, false);
    acc1 = __builtin_amdgcn_wmma_f32_16x16x32_f16(
        false, a, false, bf1, (short)0, acc1, false, false);
  }

  // BN params for this lane's 8 output channels (same for both fragments).
  float g[8], bt[8], mn[8];
  #pragma unroll
  for (int v = 0; v < 8; ++v) {
    int m = m0 + lhi * 8 + v;
    float inv = rsqrtf(var[m] + 1e-5f);
    g[v]  = gamma[m] * inv;
    bt[v] = beta[m];
    mn[v] = mean[m];
  }

  // C/D layout: lane l -> n = base + l%16 ; VGPR v -> m = m0 + (l/16)*8 + v
  int n_a = n0 + lmod;
  int n_b = n0 + 16 + lmod;
  #pragma unroll
  for (int v = 0; v < 8; ++v) {
    int m = m0 + lhi * 8 + v;
    size_t rowoff = ((size_t)(b * OC + m)) * SPAT;
    float z0 = (acc0[v] - mn[v]) * g[v] + bt[v];
    float z1 = (acc1[v] - mn[v]) * g[v] + bt[v];
    out[rowoff + n_a] = z0 / (1.0f + __expf(-z0));
    out[rowoff + n_b] = z1 / (1.0f + __expf(-z1));
  }
}

// ---------------------------------------------------------------- launch
extern "C" void kernel_launch(void* const* d_in, const int* in_sizes, int n_in,
                              void* d_out, int out_size, void* d_ws, size_t ws_size,
                              hipStream_t stream) {
  const float* x      = (const float*)d_in[0];
  const float* w1     = (const float*)d_in[1];
  const float* w2     = (const float*)d_in[2];
  const float* conv_w = (const float*)d_in[3];
  const float* gamma  = (const float*)d_in[4];
  const float* beta   = (const float*)d_in[5];
  const float* mean   = (const float*)d_in[6];
  const float* var    = (const float*)d_in[7];
  float*       out    = (float*)d_out;

  char* ws = (char*)d_ws;
  _Float16* fusedT = (_Float16*)ws;                                  // 64 MiB
  _Float16* weff   = (_Float16*)(ws + (size_t)64 * 1024 * 1024);     //  8 MiB
  float*    hfsum  = (float*)(ws + (size_t)72 * 1024 * 1024);        // 24 KB
  float*    y      = hfsum + BATCH * HFC;                            // 24 KB

  k_zero<<<(BATCH * HFC + 255) / 256, 256, 0, stream>>>(hfsum, BATCH * HFC);
  k_wavelet<<<BATCH * 8, 256, 0, stream>>>(x, fusedT, hfsum);
  k_se<<<1, 256, 0, stream>>>(hfsum, w1, w2, y);
  k_weff<<<(BATCH * OC * KC) / 512, 256, 0, stream>>>(conv_w, y, weff);
  k_gemm_bn_silu<<<BATCH * 8 * 64, 256, 0, stream>>>(
      weff, fusedT, gamma, beta, mean, var, out);
}
